// NormalToHeteroGATEncoder_37598143709582
// MI455X (gfx1250) — matrix operand order repaired and checked by reading, not compiled, not run on hardware
//
#include <hip/hip_runtime.h>
#include <hip/hip_bf16.h>

typedef __attribute__((ext_vector_type(16))) _Float16 v16h;
typedef __attribute__((ext_vector_type(8)))  float    v8f;

#define HC   512      // H*C
#define DIN  256      // K of the projections
#define DE   16       // edge feature dim
#define NHEAD 8
#define CH   64

// ---------------------------------------------------------------------------
// GEMM: P[M,512] = X[M,256] @ W[256,512], f32 in/out, f16 WMMA inside.
// Block: 256 threads (8 waves). Tile: 128(M) x 64(N), K-step 32.
// ---------------------------------------------------------------------------
__global__ __launch_bounds__(256)
void gemm_wmma_f16(const float* __restrict__ X, const float* __restrict__ W,
                   float* __restrict__ P, int M) {
    __shared__ _Float16 lds_a[128 * 40];   // [row][k], padded stride 40
    __shared__ _Float16 lds_b[64 * 40];    // [n][k] (transposed), padded

    const int t  = threadIdx.x;
    const int wv = t >> 5;                 // wave id 0..7
    const int ln = t & 31;                 // lane
    const int hf = ln >> 4;                // lane half
    const int lr = ln & 15;
    const int m0 = blockIdx.y * 128;
    const int n0 = blockIdx.x * 64;

    // A-stage row addresses are K-invariant: clamp once (rows >= M are never
    // stored, so their contents are irrelevant -- no zero-fill needed).
    const float* arow[4];
    int ac4[4];
    #pragma unroll
    for (int i = 0; i < 4; ++i) {
        int e4  = i * 256 + t;                   // 0..1023 float4 slots
        int row = e4 >> 3;                       // 8 float4 per 32-col row
        ac4[i]  = (e4 & 7) * 4;
        int rr  = m0 + row; if (rr > M - 1) rr = M - 1;
        arow[i] = X + (size_t)rr * DIN;
        // stash LDS dest row in the pointer math below via `row`
    }
    int alrow[4];
    #pragma unroll
    for (int i = 0; i < 4; ++i) alrow[i] = ((i * 256 + t) >> 3) * 40;

    v8f acc[4] = {};

    for (int k0 = 0; k0 < DIN; k0 += 32) {
        // stage A: 128x32 f32 -> f16 LDS, float4 loads
        #pragma unroll
        for (int i = 0; i < 4; ++i) {
            const float4 v = *(const float4*)(arow[i] + k0 + ac4[i]);
            _Float16* p = &lds_a[alrow[i] + ac4[i]];   // 8B-aligned (80B pitch)
            p[0] = (_Float16)v.x; p[1] = (_Float16)v.y;
            p[2] = (_Float16)v.z; p[3] = (_Float16)v.w;
        }
        // stage B transposed: 32x64 f32 -> f16 LDS [n][k], float4 loads
        #pragma unroll
        for (int i = 0; i < 2; ++i) {
            int e4   = i * 256 + t;                  // 0..511 float4 slots
            int krow = e4 >> 4;                      // 16 float4 per 64-col row
            int n4   = (e4 & 15) * 4;
            const float4 v = *(const float4*)(W + (size_t)(k0 + krow) * HC + n0 + n4);
            lds_b[(n4 + 0) * 40 + krow] = (_Float16)v.x;
            lds_b[(n4 + 1) * 40 + krow] = (_Float16)v.y;
            lds_b[(n4 + 2) * 40 + krow] = (_Float16)v.z;
            lds_b[(n4 + 3) * 40 + krow] = (_Float16)v.w;
        }
        __syncthreads();

        // A fragment: row m, K split per ISA layout
        const int m = wv * 16 + lr;
        v16h afrag;
        #pragma unroll
        for (int i = 0; i < 8; ++i) {
            afrag[i]     = lds_a[m * 40 + 8 * hf + i];        // K = 8h + i
            afrag[8 + i] = lds_a[m * 40 + 16 + 8 * hf + i];   // K = 16 + 8h + i
        }
        #pragma unroll
        for (int nt = 0; nt < 4; ++nt) {
            const int n = nt * 16 + lr;
            v16h bfrag;
            #pragma unroll
            for (int i = 0; i < 16; ++i)
                bfrag[i] = lds_b[n * 40 + 8 * hf + i + (i >= 8 ? 8 : 0)];  // K = 8h + i (i<8), 16 + 8h + (i-8) (i>=8)
            acc[nt] = __builtin_amdgcn_wmma_f32_16x16x32_f16(
                false, afrag, false, bfrag, (short)0, acc[nt], false, false);
        }
        __syncthreads();
    }

    // write back (C/D layout: VGPR r -> M = r + 8*half, N = lane%16).
    // Block-uniform branch: interior tiles take the branch-free path.
    const int mbase = m0 + wv * 16 + 8 * hf;
    const int gcolb = n0 + lr;
    if (m0 + 128 <= M) {
        #pragma unroll
        for (int nt = 0; nt < 4; ++nt) {
            float* pc = P + (size_t)mbase * HC + gcolb + nt * 16;
            #pragma unroll
            for (int r = 0; r < 8; ++r)
                pc[(size_t)r * HC] = acc[nt][r];
        }
    } else {
        #pragma unroll
        for (int nt = 0; nt < 4; ++nt) {
            #pragma unroll
            for (int r = 0; r < 8; ++r) {
                int grow = mbase + r;
                if (grow < M) P[(size_t)grow * HC + gcolb + nt * 16] = acc[nt][r];
            }
        }
    }
}

// ---------------------------------------------------------------------------
// helpers
// ---------------------------------------------------------------------------
__device__ __forceinline__ void atomicMaxF32(float* addr, float val) {
    if (!(val < 0.0f)) atomicMax((int*)addr, __float_as_int(val));
    else               atomicMin((unsigned int*)addr, __float_as_uint(val));
}

__global__ __launch_bounds__(256)
void fill_f32(float* p, float v, int n) {
    int i = blockIdx.x * 256 + threadIdx.x;
    if (i < n) p[i] = v;
}

__global__ __launch_bounds__(256)
void init_bias(float* __restrict__ out, const float* __restrict__ bias, int n) {
    int i = blockIdx.x * 256 + threadIdx.x;
    if (i < n) out[i] = bias[i & (HC - 1)];
}

// ---------------------------------------------------------------------------
// Pass 1: per-edge logits + segment max.  One block (256 thr) per edge.
// logit[e][h] = sum_c att[h*64+c] * leakyrelu(xl[src][j] + xr[dst][j] (+ ea@We))
// ---------------------------------------------------------------------------
__global__ __launch_bounds__(256)
void edge_logits_max(const float* __restrict__ xl, const float* __restrict__ xr,
                     const int* __restrict__ src, const int* __restrict__ dst,
                     const float* __restrict__ att,
                     const float* __restrict__ ea,   // [E,16] or nullptr
                     const float* __restrict__ We,   // [16,512] or nullptr
                     float* __restrict__ logits,     // [E,8]
                     float* __restrict__ mbuf) {     // [Nd,8], preset -inf
    __shared__ float red[HC];
    __shared__ float sea[DE];
    const int e = blockIdx.x;
    const int t = threadIdx.x;
    const int s = src[e];
    const int d = dst[e];
    if (ea && t < DE) sea[t] = ea[(size_t)e * DE + t];
    __syncthreads();

    #pragma unroll
    for (int part = 0; part < 2; ++part) {
        int j = t + part * 256;
        float z = xl[(size_t)s * HC + j] + xr[(size_t)d * HC + j];
        if (ea) {
            float acc = 0.0f;
            #pragma unroll
            for (int k = 0; k < DE; ++k) acc += sea[k] * We[k * HC + j];
            z += acc;
        }
        z = (z > 0.0f) ? z : 0.2f * z;     // leaky_relu(0.2)
        red[j] = z * att[j];
    }
    __syncthreads();
    // reduce 8 segments of 64
    #pragma unroll
    for (int sh = 32; sh >= 1; sh >>= 1) {
        if (t < 8 * sh) {
            int g = t / sh, o = t % sh;
            red[g * CH + o] += red[g * CH + o + sh];
        }
        __syncthreads();
    }
    if (t < NHEAD) {
        float lg = red[t * CH];
        logits[(size_t)e * NHEAD + t] = lg;
        atomicMaxF32(&mbuf[(size_t)d * NHEAD + t], lg);
    }
}

// ---------------------------------------------------------------------------
// Pass 2: e = exp(logit - max[dst]); denom[dst] += e  (elementwise over E*8)
// ---------------------------------------------------------------------------
__global__ __launch_bounds__(256)
void edge_exp_denom(const int* __restrict__ dst, int Etot,
                    float* __restrict__ logits, const float* __restrict__ mbuf,
                    float* __restrict__ denom) {
    int i = blockIdx.x * 256 + threadIdx.x;
    if (i >= Etot) return;
    int e = i >> 3, h = i & 7;
    int d = dst[e];
    float ex = __expf(logits[i] - mbuf[d * NHEAD + h]);
    logits[i] = ex;
    atomicAdd(&denom[d * NHEAD + h], ex);
}

// ---------------------------------------------------------------------------
// Pass 3: out[dst][j] += alpha[e][h] * xl[src][j].  One block per edge.
// ---------------------------------------------------------------------------
__global__ __launch_bounds__(256)
void edge_aggregate(const float* __restrict__ xl,
                    const int* __restrict__ src, const int* __restrict__ dst,
                    const float* __restrict__ evals,   // [E,8] (exp values)
                    const float* __restrict__ denom,   // [Nd,8]
                    float* __restrict__ out) {
    __shared__ float salpha[NHEAD];
    const int e = blockIdx.x;
    const int t = threadIdx.x;
    const int s = src[e];
    const int d = dst[e];
    if (t < NHEAD)
        salpha[t] = evals[(size_t)e * NHEAD + t] / (denom[d * NHEAD + t] + 1e-16f);
    __syncthreads();
    #pragma unroll
    for (int part = 0; part < 2; ++part) {
        int j = t + part * 256;
        int h = j >> 6;
        atomicAdd(&out[(size_t)d * HC + j], salpha[h] * xl[(size_t)s * HC + j]);
    }
}

// ---------------------------------------------------------------------------
extern "C" void kernel_launch(void* const* d_in, const int* in_sizes, int n_in,
                              void* d_out, int out_size, void* d_ws, size_t ws_size,
                              hipStream_t stream) {
    const float* x_user   = (const float*)d_in[0];
    const float* x_art    = (const float*)d_in[1];
    const int*   ei_f     = (const int*)d_in[2];
    const int*   ei_w     = (const int*)d_in[3];
    const float* ea_w     = (const float*)d_in[4];
    const float* Wl_f     = (const float*)d_in[5];
    const float* Wr_f     = (const float*)d_in[6];
    const float* att_f    = (const float*)d_in[7];
    const float* b_f      = (const float*)d_in[8];
    const float* Wl_w     = (const float*)d_in[9];
    const float* Wr_w     = (const float*)d_in[10];
    const float* We_w     = (const float*)d_in[11];
    const float* att_w    = (const float*)d_in[12];
    const float* b_w      = (const float*)d_in[13];

    const int NU = in_sizes[0] / DIN;
    const int NA = in_sizes[1] / DIN;
    const int EF = in_sizes[2] / 2;
    const int EW = in_sizes[3] / 2;
    const int* srcF = ei_f;        const int* dstF = ei_f + EF;
    const int* srcW = ei_w;        const int* dstW = ei_w + EW;

    float* out_user = (float*)d_out;
    float* out_art  = out_user + (size_t)NU * HC;

    // workspace carve-out (f32)
    float* ws = (float*)d_ws;
    size_t o = 0;
    float* xl_f     = ws + o; o += (size_t)NU * HC;
    float* xr_f     = ws + o; o += (size_t)NU * HC;
    float* xl_w     = ws + o; o += (size_t)NU * HC;
    float* xr_w     = ws + o; o += (size_t)NA * HC;
    float* logits_f = ws + o; o += (size_t)EF * NHEAD;
    float* logits_w = ws + o; o += (size_t)EW * NHEAD;
    float* m_f      = ws + o; o += (size_t)NU * NHEAD;
    float* den_f    = ws + o; o += (size_t)NU * NHEAD;
    float* m_w      = ws + o; o += (size_t)NA * NHEAD;
    float* den_w    = ws + o; o += (size_t)NA * NHEAD;
    (void)ws_size; (void)n_in; (void)out_size;

    // ---- dense projections (WMMA) ----
    dim3 blk(256);
    dim3 gU(HC / 64, (NU + 127) / 128);
    dim3 gA(HC / 64, (NA + 127) / 128);
    gemm_wmma_f16<<<gU, blk, 0, stream>>>(x_user, Wl_f, xl_f, NU);
    gemm_wmma_f16<<<gU, blk, 0, stream>>>(x_user, Wr_f, xr_f, NU);
    gemm_wmma_f16<<<gU, blk, 0, stream>>>(x_user, Wl_w, xl_w, NU);
    gemm_wmma_f16<<<gA, blk, 0, stream>>>(x_art,  Wr_w, xr_w, NA);

    // ---- init max/denom/output ----
    const float NEG_INF = -__builtin_inff();
    fill_f32<<<(NU * NHEAD + 255) / 256, blk, 0, stream>>>(m_f,   NEG_INF, NU * NHEAD);
    fill_f32<<<(NU * NHEAD + 255) / 256, blk, 0, stream>>>(den_f, 0.0f,    NU * NHEAD);
    fill_f32<<<(NA * NHEAD + 255) / 256, blk, 0, stream>>>(m_w,   NEG_INF, NA * NHEAD);
    fill_f32<<<(NA * NHEAD + 255) / 256, blk, 0, stream>>>(den_w, 0.0f,    NA * NHEAD);
    init_bias<<<((int)((size_t)NU * HC + 255) / 256), blk, 0, stream>>>(out_user, b_f, NU * HC);
    init_bias<<<((int)((size_t)NA * HC + 255) / 256), blk, 0, stream>>>(out_art,  b_w, NA * HC);

    // ---- relation: user -follows-> user ----
    edge_logits_max<<<EF, blk, 0, stream>>>(xl_f, xr_f, srcF, dstF, att_f,
                                            nullptr, nullptr, logits_f, m_f);
    edge_exp_denom<<<(EF * NHEAD + 255) / 256, blk, 0, stream>>>(dstF, EF * NHEAD,
                                            logits_f, m_f, den_f);
    edge_aggregate<<<EF, blk, 0, stream>>>(xl_f, srcF, dstF, logits_f, den_f, out_user);

    // ---- relation: user -writes-> article (edge features fused) ----
    edge_logits_max<<<EW, blk, 0, stream>>>(xl_w, xr_w, srcW, dstW, att_w,
                                            ea_w, We_w, logits_w, m_w);
    edge_exp_denom<<<(EW * NHEAD + 255) / 256, blk, 0, stream>>>(dstW, EW * NHEAD,
                                            logits_w, m_w, den_w);
    edge_aggregate<<<EW, blk, 0, stream>>>(xl_w, srcW, dstW, logits_w, den_w, out_art);
}
